// NasLSTM_68839735820299
// MI455X (gfx1250) — compile-verified, matrix operand order checked by reading, not run on hardware
//
#include <hip/hip_runtime.h>
#include <hip/hip_bf16.h>
#include <stdint.h>

// ---- CDNA5 WMMA operand types (gfx1250, wave32) ----
typedef __attribute__((ext_vector_type(16))) __bf16 v16bf;
typedef __attribute__((ext_vector_type(8)))  float  v8f;
typedef unsigned short u16;

union V16 { v16bf v; uint4 q[2]; };

// ---------------- fp32 -> bf16 (RNE) ----------------
__device__ __forceinline__ u16 f2bf(float f) {
    uint32_t u = __float_as_uint(f);
    u += 0x7fffu + ((u >> 16) & 1u);
    return (u16)(u >> 16);
}

__global__ void nas_f32_to_bf16(const float* __restrict__ src, u16* __restrict__ dst, int n) {
    int i = blockIdx.x * blockDim.x + threadIdx.x;
    if (i < n) dst[i] = f2bf(src[i]);
}

// ---------------- init state ----------------
__global__ void nas_init(float* h1, float* c1, float* h2, float* c2,
                         u16* h1b, u16* h2b, u16* xb) {
    int j = blockIdx.x * blockDim.x + threadIdx.x;
    if (j < 2048) { h1[j] = 0.f; c1[j] = 0.f; h2[j] = 0.f; c2[j] = 0.f; h1b[j] = 0; h2b[j] = 0; }
    if (j < 512) xb[j] = 0;
}

// ---------------- fused gates: gates = W_ih*x + b_ih + W_hh*h + b_hh -------------
// 8192 output rows. 256 threads = 8 waves/block, 16 rows/wave -> 64 blocks.
// A tile: 16 rows x 32 k of W (bf16). B tile: activation slice replicated into
// all 16 columns -> D[:,n] identical for all n; lanes 0/16 extract rows.
__global__ __launch_bounds__(256) void nas_gates_wmma(
    const u16* __restrict__ Wih, const u16* __restrict__ Whh,
    const float* __restrict__ b_ih, const float* __restrict__ b_hh,
    const u16* __restrict__ xbf, int Kx,
    const u16* __restrict__ hbf, int Kh,
    float* __restrict__ gates)
{
    const int lane = threadIdx.x & 31;
    const int wave = threadIdx.x >> 5;
    const int r0   = (blockIdx.x * 8 + wave) * 16;   // first of 16 rows for this wave
    const int rl   = lane & 15;
    const int koff = (lane < 16) ? 0 : 8;            // ISA A/B lane K-layout
    const int row  = r0 + rl;

    v8f acc = {};

    // W_ih * x
    {
        const u16* wrow = Wih + (size_t)row * Kx;
        for (int k0 = 0; k0 < Kx; k0 += 32) {
            __builtin_prefetch(wrow + k0 + 256, 0, 1);   // global_prefetch_b8
            V16 a, b;
            a.q[0] = *(const uint4*)(wrow + k0 + koff);
            a.q[1] = *(const uint4*)(wrow + k0 + koff + 16);
            b.q[0] = *(const uint4*)(xbf + k0 + koff);
            b.q[1] = *(const uint4*)(xbf + k0 + koff + 16);
            acc = __builtin_amdgcn_wmma_f32_16x16x32_bf16(
                      false, a.v, false, b.v, (short)0, acc, false, false);
        }
    }
    // W_hh * h
    {
        const u16* wrow = Whh + (size_t)row * Kh;
        for (int k0 = 0; k0 < Kh; k0 += 32) {
            __builtin_prefetch(wrow + k0 + 256, 0, 1);
            V16 a, b;
            a.q[0] = *(const uint4*)(wrow + k0 + koff);
            a.q[1] = *(const uint4*)(wrow + k0 + koff + 16);
            b.q[0] = *(const uint4*)(hbf + k0 + koff);
            b.q[1] = *(const uint4*)(hbf + k0 + koff + 16);
            acc = __builtin_amdgcn_wmma_f32_16x16x32_bf16(
                      false, a.v, false, b.v, (short)0, acc, false, false);
        }
    }

    // C/D layout: VGPR j, lane 0 -> (M=j, N=0); lane 16 -> (M=8+j, N=0)
    if (lane == 0) {
        #pragma unroll
        for (int j = 0; j < 8; ++j) { int r = r0 + j;     gates[r] = acc[j] + b_ih[r] + b_hh[r]; }
    } else if (lane == 16) {
        #pragma unroll
        for (int j = 0; j < 8; ++j) { int r = r0 + 8 + j; gates[r] = acc[j] + b_ih[r] + b_hh[r]; }
    }
}

// ---------------- LSTM cell pointwise (gate order i,f,g,o) ----------------
__device__ __forceinline__ float sigf(float x) { return 1.f / (1.f + __expf(-x)); }

__global__ void nas_cell(const float* __restrict__ gates,
                         float* __restrict__ c, float* __restrict__ h,
                         u16* __restrict__ hbf) {
    int j = blockIdx.x * blockDim.x + threadIdx.x;
    if (j >= 2048) return;
    float gi = gates[j], gf = gates[2048 + j], gg = gates[4096 + j], go = gates[6144 + j];
    float cn = sigf(gf) * c[j] + sigf(gi) * tanhf(gg);
    float hn = sigf(go) * tanhf(cn);
    c[j] = cn; h[j] = hn; hbf[j] = f2bf(hn);
}

// ---------------- head: logits, softmax, sample, fetch next embedding ----------------
__global__ __launch_bounds__(256) void nas_head(
    const float* __restrict__ h2, const float* __restrict__ W_out,
    const float* __restrict__ b_out, const float* __restrict__ emb,
    int t, u16* __restrict__ xbf, void* __restrict__ d_out)
{
    __shared__ float lg[8];
    __shared__ int sel;
    const int lane = threadIdx.x & 31;
    const int row  = threadIdx.x >> 5;                 // 8 rows, one per wave
    const float* w = W_out + ((size_t)t * 8 + row) * 2048;
    float p = 0.f;
    for (int k = lane; k < 2048; k += 32) p += h2[k] * w[k];
    for (int off = 16; off; off >>= 1) p += __shfl_down(p, off, 32);
    if (lane == 0) lg[row] = p + b_out[t * 8 + row];
    __syncthreads();

    if (threadIdx.x == 0) {
        const int sizes[4] = {8, 6, 4, 5};
        const int offs[4]  = {0, 8, 14, 18};
        int tp = t & 3, n = sizes[tp];
        float m = -1e30f;
        for (int i = 0; i < n; ++i) m = fmaxf(m, lg[i]);
        float ex[8]; float Z = 0.f;
        for (int i = 0; i < n; ++i) { ex[i] = __expf(lg[i] - m); Z += ex[i]; }
        // deterministic per-step uniform (xorshift of step index)
        uint32_t s = (uint32_t)t * 2654435761u ^ 0x9E3779B9u;
        s ^= s << 13; s ^= s >> 17; s ^= s << 5;
        float u = (float)(s >> 8) * (1.0f / 16777216.0f) * Z;
        int ind = n - 1; float cum = 0.f;
        for (int i = 0; i < n; ++i) { cum += ex[i]; if (u < cum) { ind = i; break; } }
        ((int*)d_out)[t]        = ind;            // actions (int32), first 48 slots
        ((float*)d_out)[48 + t] = ex[ind] / Z;    // action_probs
        sel = offs[tp] + ind;
    }
    __syncthreads();

    const float* er = emb + (size_t)sel * 512;
    for (int e = threadIdx.x; e < 512; e += 256) xbf[e] = f2bf(er[e]);
}

// ---------------- host orchestration ----------------
extern "C" void kernel_launch(void* const* d_in, const int* in_sizes, int n_in,
                              void* d_out, int out_size, void* d_ws, size_t ws_size,
                              hipStream_t stream) {
    (void)in_sizes; (void)n_in; (void)out_size;
    const float* emb   = (const float*)d_in[0];
    const float* W_ih1 = (const float*)d_in[1];
    const float* W_hh1 = (const float*)d_in[2];
    const float* b_ih1 = (const float*)d_in[3];
    const float* b_hh1 = (const float*)d_in[4];
    const float* W_ih2 = (const float*)d_in[5];
    const float* W_hh2 = (const float*)d_in[6];
    const float* b_ih2 = (const float*)d_in[7];
    const float* b_hh2 = (const float*)d_in[8];
    const float* W_out = (const float*)d_in[9];
    const float* b_out = (const float*)d_in[10];

    // workspace layout (all 256B aligned)
    char* ws = (char*)d_ws; size_t off = 0;
    auto alloc = [&](size_t bytes) -> char* {
        char* p = ws + off; off += (bytes + 255) & ~(size_t)255; return p;
    };
    u16* Wih1b = (u16*)alloc((size_t)8192 * 512  * 2);
    u16* Whh1b = (u16*)alloc((size_t)8192 * 2048 * 2);
    u16* Wih2b = (u16*)alloc((size_t)8192 * 2048 * 2);
    u16* Whh2b = (u16*)alloc((size_t)8192 * 2048 * 2);
    float* gates = (float*)alloc(8192 * 4);
    float* h1 = (float*)alloc(2048 * 4);
    float* c1 = (float*)alloc(2048 * 4);
    float* h2 = (float*)alloc(2048 * 4);
    float* c2 = (float*)alloc(2048 * 4);
    u16* h1b = (u16*)alloc(2048 * 2);
    u16* h2b = (u16*)alloc(2048 * 2);
    u16* xb  = (u16*)alloc(512 * 2);
    if (off > ws_size) return;   // deterministic no-op if scratch too small

    // One-time (per launch) bf16 weight packing: 208MB fp32 -> 104MB bf16,
    // which fits in the 192MB L2 so the 48-step loop runs L2-resident.
    int n;
    n = 8192 * 512;  nas_f32_to_bf16<<<(n + 255) / 256, 256, 0, stream>>>(W_ih1, Wih1b, n);
    n = 8192 * 2048; nas_f32_to_bf16<<<(n + 255) / 256, 256, 0, stream>>>(W_hh1, Whh1b, n);
    nas_f32_to_bf16<<<(n + 255) / 256, 256, 0, stream>>>(W_ih2, Wih2b, n);
    nas_f32_to_bf16<<<(n + 255) / 256, 256, 0, stream>>>(W_hh2, Whh2b, n);
    nas_init<<<8, 256, 0, stream>>>(h1, c1, h2, c2, h1b, h2b, xb);

    for (int t = 0; t < 48; ++t) {
        nas_gates_wmma<<<64, 256, 0, stream>>>(Wih1b, Whh1b, b_ih1, b_hh1,
                                               xb, 512, h1b, 2048, gates);
        nas_cell<<<8, 256, 0, stream>>>(gates, c1, h1, h1b);
        nas_gates_wmma<<<64, 256, 0, stream>>>(Wih2b, Whh2b, b_ih2, b_hh2,
                                               h1b, 2048, h2b, 2048, gates);
        nas_cell<<<8, 256, 0, stream>>>(gates, c2, h2, h2b);
        nas_head<<<1, 256, 0, stream>>>(h2, W_out, b_out, emb, t, xb, d_out);
    }
}
